// MOE_56281251447395
// MI455X (gfx1250) — compile-verified, matrix operand order checked by reading, not run on hardware
//
#include <hip/hip_runtime.h>
#include <hip/hip_bf16.h>

// MoE: B=8, L=8192 (T=65536 tokens), D=512, H=2048, E=8, K=2 (top-k)
#define Tn 65536
#define Dp 512
#define Hp 2048
#define Ep 8

typedef __attribute__((ext_vector_type(16))) __bf16          v16bf;
typedef __attribute__((ext_vector_type(16))) unsigned short  v16us;
typedef __attribute__((ext_vector_type(8)))  unsigned short  v8us;
typedef __attribute__((ext_vector_type(8)))  float           v8f;

__device__ __forceinline__ unsigned short f32_to_bf16(float f) {
  unsigned u = __float_as_uint(f);
  unsigned r = u + 0x7FFFu + ((u >> 16) & 1u);   // round-to-nearest-even
  return (unsigned short)(r >> 16);
}

// Combine two contiguous 16B LDS vectors into a v16bf WMMA operand.
__device__ __forceinline__ v16bf frag_from(const unsigned short* p) {
  v8us lo = *(const v8us*)p;
  v8us hi = *(const v8us*)(p + 8);
  v16us t = __builtin_shufflevector(lo, hi, 0,1,2,3,4,5,6,7,8,9,10,11,12,13,14,15);
  return __builtin_bit_cast(v16bf, t);
}

// A-matrix fragment (16x32 bf16) from row-major LDS buffer, per ISA 7.12.2:
// lane&15 = M row, lane<16 -> K {0..7,16..23}, lane>=16 -> K {8..15,24..31}.
__device__ __forceinline__ v16bf frag_A(const unsigned short* buf, int row0,
                                        int k0, int stride, int lane) {
  int m   = lane & 15;
  int sel = (lane >> 4) << 3;                    // 0 or 8
  const unsigned short* p = buf + (row0 + m) * stride + k0 + sel;
  v8us lo = *(const v8us*)p;                     // K = sel .. sel+7
  v8us hi = *(const v8us*)(p + 16);              // K = 16+sel .. 16+sel+7
  v16us t = __builtin_shufflevector(lo, hi, 0,1,2,3,4,5,6,7,8,9,10,11,12,13,14,15);
  return __builtin_bit_cast(v16bf, t);
}

// ---------------- weight fp32 -> bf16 with WMMA B-fragment swizzle ----------------
// Output layout: [e][tile][lane(32)][16], tile = nt*Tk+kt (kMajor=0) or kt*Tn+nt (kMajor=1).
// Lane l holds, for column n = nt*16+(l&15), K values {s..s+7} U {16+s..16+s+7}, s=8*(l>>4).
__global__ __launch_bounds__(256) void swizzle_b_kernel(const float* __restrict__ in,
                                                        unsigned short* __restrict__ out,
                                                        int K, int N, int kMajor) {
  int Tk = K >> 5, Tn2 = N >> 4;
  int tilesPerE = Tk * Tn2;
  int wave = (int)((blockIdx.x * blockDim.x + threadIdx.x) >> 5);
  int lane = threadIdx.x & 31;
  int e  = wave / tilesPerE;
  if (e >= Ep) return;
  int tl = wave % tilesPerE;
  int kt, nt;
  if (kMajor) { kt = tl / Tn2; nt = tl % Tn2; }
  else        { nt = tl / Tk;  kt = tl % Tk;  }
  const float* src = in + (size_t)e * K * N;
  int n     = nt * 16 + (lane & 15);
  int kbase = kt * 32 + ((lane >> 4) << 3);
  v8us lo, hi;
#pragma unroll
  for (int i = 0; i < 8; ++i) {
    lo[i] = f32_to_bf16(src[(size_t)(kbase + i)      * N + n]);
    hi[i] = f32_to_bf16(src[(size_t)(kbase + 16 + i) * N + n]);
  }
  unsigned short* dst = out + ((size_t)(e * tilesPerE + tl) * 32 + lane) * 16;
  *(v8us*)dst       = lo;
  *(v8us*)(dst + 8) = hi;
}

// ---------------- gating: softmax -> top-2 -> renorm -> route ----------------
__global__ __launch_bounds__(256) void moe_gate_kernel(const float* __restrict__ x,
                                                       const float* __restrict__ Wg,
                                                       int* __restrict__ cnt,
                                                       int* __restrict__ tokIdx,
                                                       float* __restrict__ tokW) {
  int tok  = (int)((blockIdx.x * blockDim.x + threadIdx.x) >> 5); // one wave32 per token
  int lane = threadIdx.x & 31;
  if (tok >= Tn) return;
  const float* xr = x + (size_t)tok * Dp;

  float acc[Ep] = {};
#pragma unroll 4
  for (int j = 0; j < Dp / 32; ++j) {
    int d = lane + 32 * j;
    float xv = xr[d];
    const float* wrow = Wg + d * Ep;
#pragma unroll
    for (int e = 0; e < Ep; ++e) acc[e] += xv * wrow[e];
  }
#pragma unroll
  for (int e = 0; e < Ep; ++e) {
    float v = acc[e];
    for (int off = 16; off; off >>= 1) v += __shfl_xor(v, off, 32);
    acc[e] = v;
  }
  float mx = acc[0];
#pragma unroll
  for (int e = 1; e < Ep; ++e) mx = fmaxf(mx, acc[e]);
  float s = 0.0f;
#pragma unroll
  for (int e = 0; e < Ep; ++e) { acc[e] = expf(acc[e] - mx); s += acc[e]; }
  float inv = 1.0f / s;
#pragma unroll
  for (int e = 0; e < Ep; ++e) acc[e] *= inv;
  int i1 = 0; float v1 = acc[0];
#pragma unroll
  for (int e = 1; e < Ep; ++e) if (acc[e] > v1) { v1 = acc[e]; i1 = e; }
  int i2 = (i1 == 0) ? 1 : 0; float v2 = acc[i2];
#pragma unroll
  for (int e = 0; e < Ep; ++e) if (e != i1 && acc[e] > v2) { v2 = acc[e]; i2 = e; }
  float denom = v1 + v2 + 1e-6f;

  if (lane == 0) {
    int s1 = atomicAdd(&cnt[i1], 1);
    tokIdx[i1 * Tn + s1] = tok;  tokW[i1 * Tn + s1] = v1 / denom;
    int s2 = atomicAdd(&cnt[i2], 1);
    tokIdx[i2 * Tn + s2] = tok;  tokW[i2 * Tn + s2] = v2 / denom;
  }
}

// ---------------- fused expert FFN: gelu(X@W1+b1)@W2+b2, scaled scatter-add ----------------
__global__ __launch_bounds__(256) void moe_ffn_kernel(
    const float* __restrict__ x,
    const unsigned short* __restrict__ w1sw,   // swizzled [e][nt(128)][kt(16)][lane][16]
    const unsigned short* __restrict__ w2sw,   // swizzled [e][kt(64)][nt(32)][lane][16]
    const float* __restrict__ b1,
    const float* __restrict__ b2,
    const int* __restrict__ cnt,
    const int* __restrict__ tokIdx,
    const float* __restrict__ tokW,
    float* __restrict__ out) {
  __shared__ v8us sW1v[2048];                                        // 32 KB: chunk, frag layout
  __shared__ v8us sW2v[2048];                                        // 32 KB: chunk, frag layout
  __shared__ unsigned short sX[64 * 520] __attribute__((aligned(16))); // 66.5 KB, row-major
  __shared__ unsigned short sHcA[4 * 32 * 16] __attribute__((aligned(16))); // 4 KB, A-frag layout
  __shared__ int   sTok[64];
  __shared__ float sWt[64];

  const int e  = blockIdx.y;
  const int m0 = blockIdx.x * 64;
  const int cn = cnt[e];
  if (m0 >= cn) return;
  const int rows = min(64, cn - m0);
  const int tid  = threadIdx.x;
  const int lane = tid & 31;
  const int w    = tid >> 5;                 // wave id 0..7
  const int mt    = w & 3;                   // wave's M tile (16 rows)
  const int mrow  = mt * 16;
  const int nhalf = w >> 2;                  // wave's N half (0/1) of 512 cols
  const unsigned short* sW1 = (const unsigned short*)sW1v;
  const unsigned short* sW2 = (const unsigned short*)sW2v;

  // warm the cache for the first weight chunk (global_prefetch, no VGPR/counter cost)
  {
    const char* c1 = (const char*)(w1sw + ((size_t)e * 2048) * 512);
    const char* c2 = (const char*)(w2sw + ((size_t)e * 2048) * 512);
    __builtin_prefetch(c1 + tid * 128, 0, 3);
    __builtin_prefetch(c2 + tid * 128, 0, 3);
  }

  if (tid < 64) {
    if (tid < rows) { sTok[tid] = tokIdx[e * Tn + m0 + tid]; sWt[tid] = tokW[e * Tn + m0 + tid]; }
    else            { sTok[tid] = tokIdx[e * Tn + m0];       sWt[tid] = 0.0f; }
  }
  __syncthreads();

  // ---- stage X tile: fp32 global -> bf16 LDS (float4 reads, 8B packed stores) ----
  for (int idx = tid; idx < 64 * (Dp / 4); idx += 256) {
    int r = idx >> 7, d4 = idx & 127;
    float4 v;
    if (r < rows) v = *(const float4*)(x + (size_t)sTok[r] * Dp + d4 * 4);
    else          v = make_float4(0.f, 0.f, 0.f, 0.f);
    unsigned long long pk =
        (unsigned long long)f32_to_bf16(v.x)
      | ((unsigned long long)f32_to_bf16(v.y) << 16)
      | ((unsigned long long)f32_to_bf16(v.z) << 32)
      | ((unsigned long long)f32_to_bf16(v.w) << 48);
    *(unsigned long long*)(sX + r * 520 + d4 * 4) = pk;
  }

  const v8f zero = {0.f, 0.f, 0.f, 0.f, 0.f, 0.f, 0.f, 0.f};
  v8f acc[16];
#pragma unroll
  for (int t = 0; t < 16; ++t) acc[t] = zero;

  const float* b1e = b1 + e * Hp;

  for (int hc = 0; hc < Hp; hc += 32) {
    const v8us* g1 = (const v8us*)(w1sw + ((size_t)e * 2048 + (hc >> 4) * 16) * 512);
    const v8us* g2 = (const v8us*)(w2sw + ((size_t)e * 2048 + (hc >> 5) * 32) * 512);
    __syncthreads();                         // previous chunk's readers done
    // stage weight chunk (hits WGP$/L2 thanks to last iteration's prefetch)
#pragma unroll
    for (int j = 0; j < 8; ++j) sW1v[tid + 256 * j] = g1[tid + 256 * j];
#pragma unroll
    for (int j = 0; j < 8; ++j) sW2v[tid + 256 * j] = g2[tid + 256 * j];
    // prefetch next chunk into cache (fire-and-forget)
    if (hc + 32 < Hp) {
      int hn = hc + 32;
      const char* c1 = (const char*)(w1sw + ((size_t)e * 2048 + (hn >> 4) * 16) * 512);
      const char* c2 = (const char*)(w2sw + ((size_t)e * 2048 + (hn >> 5) * 32) * 512);
      __builtin_prefetch(c1 + tid * 128, 0, 3);
      __builtin_prefetch(c2 + tid * 128, 0, 3);
    }
    __syncthreads();

    // ---- Stage A: Hc[64x32] = gelu(X @ W1c + b1); one 16x16 tile per wave ----
    // Two independent operand buffers (aA/bA, aB/bB) force the allocator to keep
    // the next iteration's ds_load_b128s in flight under the current WMMA.
    v8f ha = zero;
    {
      v16bf aA = frag_A(sX, mrow, 0,  520, lane);
      v16bf bA = frag_from(sW1 + ((nhalf * 16 + 0) * 32 + lane) * 16);
      v16bf aB = frag_A(sX, mrow, 32, 520, lane);
      v16bf bB = frag_from(sW1 + ((nhalf * 16 + 1) * 32 + lane) * 16);
#pragma unroll
      for (int kt = 0; kt < 16; kt += 2) {
        ha = __builtin_amdgcn_wmma_f32_16x16x32_bf16(false, aA, false, bA, (short)0, ha, false, false);
        if (kt + 2 < 16) {
          aA = frag_A(sX, mrow, (kt + 2) * 32, 520, lane);
          bA = frag_from(sW1 + ((nhalf * 16 + kt + 2) * 32 + lane) * 16);
        }
        ha = __builtin_amdgcn_wmma_f32_16x16x32_bf16(false, aB, false, bB, (short)0, ha, false, false);
        if (kt + 3 < 16) {
          aB = frag_A(sX, mrow, (kt + 3) * 32, 520, lane);
          bB = frag_from(sW1 + ((nhalf * 16 + kt + 3) * 32 + lane) * 16);
        }
      }
    }
    {
      int col   = nhalf * 16 + (lane & 15);          // 0..31 within chunk
      int rbase = mrow + ((lane >> 4) << 3);         // C/D layout: lanes>=16 hold M+8
      float bias = b1e[hc + col];
      int lane2 = ((col >> 3) & 1) << 4;             // which lane half holds this K
      int i2    = (col & 7) + ((col >> 4) << 3);     // element index within fragment
#pragma unroll
      for (int r = 0; r < 8; ++r) {
        float v = ha[r] + bias;
        v = 0.5f * v * (1.0f + erff(v * 0.70710678118654752f)); // exact GELU
        int mprime = (rbase + r) & 15;
        sHcA[(mt * 32 + (lane2 + mprime)) * 16 + i2] = f32_to_bf16(v);
      }
    }
    __syncthreads();

    // ---- Stage B: Y[64x512] += Hc @ W2c; each wave: 16 rows x 256 cols ----
    {
      v16bf a  = frag_from(sHcA + (mt * 32 + lane) * 16);
      v16bf bA = frag_from(sW2 + ((nhalf * 16 + 0) * 32 + lane) * 16);
      v16bf bB = frag_from(sW2 + ((nhalf * 16 + 1) * 32 + lane) * 16);
#pragma unroll
      for (int t = 0; t < 16; t += 2) {
        acc[t] = __builtin_amdgcn_wmma_f32_16x16x32_bf16(false, a, false, bA, (short)0, acc[t], false, false);
        if (t + 2 < 16)
          bA = frag_from(sW2 + ((nhalf * 16 + t + 2) * 32 + lane) * 16);
        acc[t + 1] = __builtin_amdgcn_wmma_f32_16x16x32_bf16(false, a, false, bB, (short)0, acc[t + 1], false, false);
        if (t + 3 < 16)
          bB = frag_from(sW2 + ((nhalf * 16 + t + 3) * 32 + lane) * 16);
      }
    }
  }

  // ---- (Y + b2) * gate_weight -> scatter-add (token is in 2 expert lists) ----
  {
    const float* b2e = b2 + e * Dp;
    int ncol  = lane & 15;
    int rbase = mrow + ((lane >> 4) << 3);
#pragma unroll
    for (int t = 0; t < 16; ++t) {
      int col = nhalf * 256 + t * 16 + ncol;
      float bias = b2e[col];
#pragma unroll
      for (int r = 0; r < 8; ++r) {
        int row = rbase + r;
        float val = (acc[t][r] + bias) * sWt[row];
        unsafeAtomicAdd(out + (size_t)sTok[row] * Dp + col, val);
      }
    }
  }
}

extern "C" void kernel_launch(void* const* d_in, const int* in_sizes, int n_in,
                              void* d_out, int out_size, void* d_ws, size_t ws_size,
                              hipStream_t stream) {
  (void)in_sizes; (void)n_in; (void)ws_size;
  const float* x  = (const float*)d_in[0];
  const float* Wg = (const float*)d_in[1];
  const float* W1 = (const float*)d_in[2];
  const float* b1 = (const float*)d_in[3];
  const float* W2 = (const float*)d_in[4];
  const float* b2 = (const float*)d_in[5];
  float* out = (float*)d_out;

  const size_t W_ELEMS = (size_t)Ep * Dp * Hp;          // 8,388,608 per weight tensor
  char* ws = (char*)d_ws;
  unsigned short* w1sw = (unsigned short*)ws;
  unsigned short* w2sw = w1sw + W_ELEMS;
  char* p = ws + 2 * W_ELEMS * sizeof(unsigned short);
  int*   cntp   = (int*)p;   p += 256;
  int*   tokIdx = (int*)p;   p += (size_t)Ep * Tn * sizeof(int);
  float* tokW   = (float*)p;

  hipMemsetAsync(d_out, 0, (size_t)out_size * sizeof(float), stream);
  hipMemsetAsync(cntp, 0, 256, stream);

  // 8 waves/block; one wave per 32x16 tile: E*Tk*Tn waves = 16384 -> 2048 blocks
  hipLaunchKernelGGL(swizzle_b_kernel, dim3(2048), dim3(256), 0, stream,
                     W1, w1sw, Dp, Hp, /*kMajor=*/0);
  hipLaunchKernelGGL(swizzle_b_kernel, dim3(2048), dim3(256), 0, stream,
                     W2, w2sw, Hp, Dp, /*kMajor=*/1);
  hipLaunchKernelGGL(moe_gate_kernel, dim3(Tn / 8), dim3(256), 0, stream,
                     x, Wg, cntp, tokIdx, tokW);
  hipLaunchKernelGGL(moe_ffn_kernel, dim3(Tn / 64, Ep), dim3(256), 0, stream,
                     x, w1sw, w2sw, b1, b2, cntp, tokIdx, tokW, out);
}
